// GAT_3255585210651
// MI455X (gfx1250) — matrix operand order repaired and checked by reading, not compiled, not run on hardware
//
#include <hip/hip_runtime.h>
#include <hip/hip_bf16.h>
#include <math.h>

typedef __attribute__((ext_vector_type(16))) __bf16 v16bf;
typedef __attribute__((ext_vector_type(8)))  float  v8f;

#define NEG_SLOPE 0.2f

// ---------- helpers ----------
__device__ __forceinline__ unsigned short f2bf(float f) {
    unsigned int u = __float_as_uint(f);
    unsigned int r = (u + 0x7FFFu + ((u >> 16) & 1u)) >> 16;   // RNE
    return (unsigned short)r;
}
// monotonic float<->uint mapping for atomic max on floats (incl. negatives)
__device__ __forceinline__ unsigned int encf(float x) {
    unsigned int u = __float_as_uint(x);
    return (u & 0x80000000u) ? ~u : (u | 0x80000000u);
}
__device__ __forceinline__ float decf(unsigned int u) {
    unsigned int b = (u & 0x80000000u) ? (u ^ 0x80000000u) : ~u;
    return __uint_as_float(b);
}

union FragBF {
    v16bf v;
    unsigned short u[16];
    uint4 q[2];
};

// ---------- 1) argmax over 128 features + embedding gather -> bf16 padded rows ----------
__global__ void k_argmax_embed(const float* __restrict__ x, const float* __restrict__ emb,
                               unsigned short* __restrict__ hbf, int N, int V, int D) {
    int wave = blockIdx.x * (blockDim.x >> 5) + (threadIdx.x >> 5);
    if (wave >= N) return;
    int lane = threadIdx.x & 31;
    const float* xr = x + (size_t)wave * V;
    float best = -INFINITY; int bi = 0x7fffffff;
    for (int k = lane; k < V; k += 32) {
        float v = xr[k];
        if (v > best || (v == best && k < bi)) { best = v; bi = k; }
    }
    #pragma unroll
    for (int m = 16; m >= 1; m >>= 1) {
        float ov = __shfl_xor(best, m, 32);
        int   oi = __shfl_xor(bi, m, 32);
        if (ov > best || (ov == best && oi < bi)) { best = ov; bi = oi; }
    }
    const float* er = emb + (size_t)bi * D;
    unsigned short* hr = hbf + (size_t)wave * 64;
    for (int c = lane; c < 64; c += 32)
        hr[c] = (c < D) ? f2bf(er[c]) : (unsigned short)0;
}

// ---------- 2) WMMA GEMM: hout = hbf @ W ; alpha_src/dst per node ----------
// W is staged once per block into LDS, bf16, zero-padded, transposed to [n][kPad]
// so a lane's B fragment = two contiguous ds 16B loads.
template<int KSTEPS>
__global__ void k_gemm_gat(const unsigned short* __restrict__ hbf,  // [N,64] bf16, zero padded
                           const float* __restrict__ W,             // [Fin,Fout] f32
                           const float* __restrict__ a_src, const float* __restrict__ a_dst,
                           float* __restrict__ hout,                // [N,Fout]
                           float* __restrict__ asrc, float* __restrict__ adst,
                           int N, int Fin, int Fout) {
    constexpr int KPAD = 64;                       // covers KSTEPS<=2
    __shared__ __align__(16) unsigned short wlds[16 * KPAD];   // [n][k], 2 KB

    // cooperative fill (uniform across block, before any early-out)
    for (int i = threadIdx.x; i < 16 * KPAD; i += blockDim.x) {
        int n = i >> 6, k = i & (KPAD - 1);
        float w = (k < Fin && n < Fout) ? W[(size_t)k * Fout + n] : 0.0f;
        wlds[i] = f2bf(w);
    }
    __syncthreads();

    int wave = blockIdx.x * (blockDim.x >> 5) + (threadIdx.x >> 5);
    int row0 = wave * 16;
    if (row0 >= N) return;                 // wave-uniform
    int lane = threadIdx.x & 31;
    int half = lane >> 4;                  // 0 or 1
    int l16  = lane & 15;

    int arow = row0 + l16;                 // A: both halves carry M=0..15
    if (arow >= N) arow = N - 1;           // clamp (tail safety; results masked below)
    const unsigned short* ap = hbf + (size_t)arow * 64;

    v8f c = {};
    #pragma unroll
    for (int ks = 0; ks < KSTEPS; ++ks) {
        FragBF a, b;
        // A layout (16-bit, 16x32): lanes0-15 K=0..7,16..23 ; lanes16-31 K=8..15,24..31
        int ka = ks * 32 + half * 8;
        a.q[0] = *(const uint4*)(ap + ka);
        a.q[1] = *(const uint4*)(ap + ka + 16);
        // B layout (16-bit, 32x16): N = lane&15 ; lanes0-15 K=0..7,16..23 ; lanes16-31 K=8..15,24..31
        int kb = ks * 32 + half * 8;
        const unsigned short* wp = &wlds[l16 * KPAD + kb];
        b.q[0] = *(const uint4*)(wp);
        b.q[1] = *(const uint4*)(wp + 16);
        c = __builtin_amdgcn_wmma_f32_16x16x32_bf16(false, a.v, false, b.v,
                                                    (short)0, c, false, false);
    }

    // C/D layout: VGPR r -> M = r + 8*half ; N = lane&15
    float as_w = (l16 < Fout) ? a_src[l16] : 0.0f;
    float ad_w = (l16 < Fout) ? a_dst[l16] : 0.0f;
    #pragma unroll
    for (int r = 0; r < 8; ++r) {
        int node = row0 + r + half * 8;
        float v = c[r];
        bool ok = (node < N);
        if (ok && l16 < Fout) hout[(size_t)node * Fout + l16] = v;
        float ps = v * as_w, pd = v * ad_w;
        #pragma unroll
        for (int m = 8; m >= 1; m >>= 1) {     // stays inside the 16-lane half
            ps += __shfl_xor(ps, m, 32);
            pd += __shfl_xor(pd, m, 32);
        }
        if (ok && l16 == 0) { asrc[node] = ps; adst[node] = pd; }
    }
}

// ---------- 3) per-layer init ----------
__global__ void k_init_layer(unsigned int* __restrict__ mx, float* __restrict__ den,
                             float* __restrict__ agg, int N, int NF) {
    int i = blockIdx.x * blockDim.x + threadIdx.x;
    if (i < N) { mx[i] = 0x007FFFFFu; /* enc(-inf) */ den[i] = 0.0f; }
    if (i < NF) agg[i] = 0.0f;
}

// ---------- 4) edge pass A: e = leaky_relu(asrc[s]+adst[d]); segment max ----------
__global__ void k_edge_a(const int* __restrict__ src, const int* __restrict__ dst, int E, int N,
                         const float* __restrict__ asrc, const float* __restrict__ adst,
                         float* __restrict__ e_buf, unsigned int* __restrict__ mx) {
    int eid = blockIdx.x * blockDim.x + threadIdx.x;
    int ET = E + N;
    if (eid >= ET) return;
    int s = (eid < E) ? src[eid] : (eid - E);
    int d = (eid < E) ? dst[eid] : (eid - E);
    float v = asrc[s] + adst[d];
    v = (v > 0.0f) ? v : NEG_SLOPE * v;
    e_buf[eid] = v;
    atomicMax(&mx[d], encf(v));
}

// ---------- 5) edge pass B: p = exp(e - m[d]); segment sum -> den ----------
__global__ void k_edge_b(const int* __restrict__ dst, int E, int N,
                         const float* __restrict__ e_buf, const unsigned int* __restrict__ mx,
                         float* __restrict__ den) {
    int eid = blockIdx.x * blockDim.x + threadIdx.x;
    int ET = E + N;
    if (eid >= ET) return;
    int d = (eid < E) ? dst[eid] : (eid - E);
    float p = __expf(e_buf[eid] - decf(mx[d]));
    atomicAdd(&den[d], p);
}

// ---------- 6) edge pass C: agg[d] += (p/den[d]) * hout[s] ; 16 lanes per edge ----------
__global__ void k_edge_c(const int* __restrict__ src, const int* __restrict__ dst, int E, int N,
                         const float* __restrict__ e_buf, const unsigned int* __restrict__ mx,
                         const float* __restrict__ den, const float* __restrict__ hout,
                         float* __restrict__ agg, int Fout) {
    int t = blockIdx.x * blockDim.x + threadIdx.x;
    int eid = t >> 4;
    int f   = t & 15;
    int ET = E + N;
    if (eid >= ET) return;
    int s = (eid < E) ? src[eid] : (eid - E);
    int d = (eid < E) ? dst[eid] : (eid - E);
    float p  = __expf(e_buf[eid] - decf(mx[d]));
    float dn = den[d]; dn = (dn > 1e-16f) ? dn : 1e-16f;
    float coef = p / dn;
    if (f < Fout)
        atomicAdd(&agg[(size_t)d * Fout + f], coef * hout[(size_t)s * Fout + f]);
}

// ---------- 7a) finalize (layers 1,2): bias + relu -> bf16 padded rows ----------
__global__ void k_finalize_mid(const float* __restrict__ agg, const float* __restrict__ bias,
                               unsigned short* __restrict__ hbf, int N, int Fout) {
    int i = blockIdx.x * blockDim.x + threadIdx.x;
    if (i >= N) return;
    unsigned short* hr = hbf + (size_t)i * 64;
    #pragma unroll
    for (int f = 0; f < 32; ++f) {
        float v = 0.0f;
        if (f < Fout) {
            v = agg[(size_t)i * Fout + f] + bias[f];
            v = (v > 0.0f) ? v : 0.0f;
        }
        hr[f] = (f < Fout) ? f2bf(v) : (unsigned short)0;
    }
}

// ---------- 7b) finalize (layer 3): bias, pool into per-graph sums ----------
__global__ void k_zero_pool(float* __restrict__ pooled, float* __restrict__ cnt, int G, int C) {
    int i = blockIdx.x * blockDim.x + threadIdx.x;
    if (i < G * C) pooled[i] = 0.0f;
    if (i < G) cnt[i] = 0.0f;
}

__global__ void k_finalize_last(const float* __restrict__ agg, const float* __restrict__ bias,
                                const int* __restrict__ batch, float* __restrict__ pooled,
                                float* __restrict__ cnt, int N, int C) {
    int i = blockIdx.x * blockDim.x + threadIdx.x;
    if (i >= N) return;
    int g = batch[i];
    for (int c = 0; c < C; ++c)
        atomicAdd(&pooled[(size_t)g * C + c], agg[(size_t)i * C + c] + bias[c]);
    atomicAdd(&cnt[g], 1.0f);
}

// ---------- 8) mean + softmax per graph ----------
__global__ void k_pool_softmax(const float* __restrict__ pooled, const float* __restrict__ cnt,
                               float* __restrict__ out, int G, int C) {
    int g = blockIdx.x * blockDim.x + threadIdx.x;
    if (g >= G) return;
    float n = cnt[g]; n = (n > 1.0f) ? n : 1.0f;
    float vals[16];
    float mx = -INFINITY;
    for (int c = 0; c < C; ++c) {
        float v = pooled[(size_t)g * C + c] / n;
        vals[c] = v;
        if (v > mx) mx = v;
    }
    float s = 0.0f;
    for (int c = 0; c < C; ++c) { float e = __expf(vals[c] - mx); vals[c] = e; s += e; }
    for (int c = 0; c < C; ++c) out[(size_t)g * C + c] = vals[c] / s;
}

// =========================================================================
extern "C" void kernel_launch(void* const* d_in, const int* in_sizes, int n_in,
                              void* d_out, int out_size, void* d_ws, size_t ws_size,
                              hipStream_t stream) {
    const float* x   = (const float*)d_in[0];
    const int*   ei  = (const int*)  d_in[1];
    const int*   bat = (const int*)  d_in[2];
    const float* emb = (const float*)d_in[3];

    const float* Wl[3]  = { (const float*)d_in[4],  (const float*)d_in[8],  (const float*)d_in[12] };
    const float* asl[3] = { (const float*)d_in[5],  (const float*)d_in[9],  (const float*)d_in[13] };
    const float* adl[3] = { (const float*)d_in[6],  (const float*)d_in[10], (const float*)d_in[14] };
    const float* bl[3]  = { (const float*)d_in[7],  (const float*)d_in[11], (const float*)d_in[15] };

    const int N = in_sizes[2];             // batch is [N]
    const int V = in_sizes[0] / N;         // 128
    const int E = in_sizes[1] / 2;
    const int D = in_sizes[3] / V;         // 50
    const int H = in_sizes[5];             // 16
    const int C = in_sizes[13];            // 10
    const int G = out_size / C;            // 512
    const int ET = E + N;

    // workspace layout (256B aligned)
    size_t off = 0;
    auto wsa = [&](size_t bytes) { size_t r = off; off = (off + bytes + 255) & ~(size_t)255; return r; };
    char* ws = (char*)d_ws;
    unsigned short* hbf   = (unsigned short*)(ws + wsa((size_t)N * 64 * 2));
    float*          hout  = (float*)        (ws + wsa((size_t)N * 16 * 4));
    float*          agg   = (float*)        (ws + wsa((size_t)N * 16 * 4));
    float*          asrc  = (float*)        (ws + wsa((size_t)N * 4));
    float*          adst  = (float*)        (ws + wsa((size_t)N * 4));
    float*          e_buf = (float*)        (ws + wsa((size_t)ET * 4));
    unsigned int*   mx    = (unsigned int*) (ws + wsa((size_t)N * 4));
    float*          den   = (float*)        (ws + wsa((size_t)N * 4));
    float*          pooled= (float*)        (ws + wsa((size_t)G * 16 * 4));
    float*          cnt   = (float*)        (ws + wsa((size_t)G * 4));
    (void)ws_size;

    const int* srcArr = ei;        // edge_index[0]
    const int* dstArr = ei + E;    // edge_index[1]

    const int TB = 256;
    const int wavesPerBlock = TB / 32;

    // 1) argmax + embedding gather (one wave per node)
    {
        int blocks = (N + wavesPerBlock - 1) / wavesPerBlock;
        k_argmax_embed<<<blocks, TB, 0, stream>>>(x, emb, hbf, N, V, D);
    }

    const int FinL[3]  = { D, H, H };
    const int FoutL[3] = { H, H, C };

    for (int L = 0; L < 3; ++L) {
        int Fin = FinL[L], Fout = FoutL[L];

        k_init_layer<<<((size_t)N * Fout + TB - 1) / TB, TB, 0, stream>>>(mx, den, agg, N, N * Fout);

        int tiles = (N + 15) / 16;
        int gblocks = (tiles + wavesPerBlock - 1) / wavesPerBlock;
        if (Fin > 32) {
            k_gemm_gat<2><<<gblocks, TB, 0, stream>>>(
                hbf, Wl[L], asl[L], adl[L], hout, asrc, adst, N, Fin, Fout);
        } else {
            k_gemm_gat<1><<<gblocks, TB, 0, stream>>>(
                hbf, Wl[L], asl[L], adl[L], hout, asrc, adst, N, Fin, Fout);
        }

        k_edge_a<<<(ET + TB - 1) / TB, TB, 0, stream>>>(srcArr, dstArr, E, N, asrc, adst, e_buf, mx);
        k_edge_b<<<(ET + TB - 1) / TB, TB, 0, stream>>>(dstArr, E, N, e_buf, mx, den);
        {
            long long tot = (long long)ET * 16;
            k_edge_c<<<(unsigned int)((tot + TB - 1) / TB), TB, 0, stream>>>(
                srcArr, dstArr, E, N, e_buf, mx, den, hout, agg, Fout);
        }

        if (L < 2) {
            k_finalize_mid<<<(N + TB - 1) / TB, TB, 0, stream>>>(agg, bl[L], hbf, N, Fout);
        } else {
            k_zero_pool<<<((size_t)G * C + TB - 1) / TB, TB, 0, stream>>>(pooled, cnt, G, C);
            k_finalize_last<<<(N + TB - 1) / TB, TB, 0, stream>>>(agg, bl[L], bat, pooled, cnt, N, C);
        }
    }

    k_pool_softmax<<<(G + TB - 1) / TB, TB, 0, stream>>>(pooled, cnt, (float*)d_out, G, C);
}
